// MultiWayTransformer_84258668413381
// MI455X (gfx1250) — matrix operand order repaired and checked by reading, not compiled, not run on hardware
//
#include <hip/hip_runtime.h>
#include <hip/hip_bf16.h>

// ---------------- problem dims ----------------
constexpr int Lq   = 12;
constexpr int Bb   = 4;
constexpr int Nn   = 1024;
constexpr int Dm   = 768;
constexpr int Hh   = 12;
constexpr int DHd  = 64;       // Dm / Hh
constexpr int HIDd = 3072;     // 4*Dm
constexpr int Rr   = 2050;
constexpr int Tt   = 512;
constexpr int Mrows = Bb * Nn; // 4096
constexpr float SCALE = 0.125f; // DH^-0.5

// ---------------- WMMA types ----------------
typedef __bf16 bf16_t;
typedef __attribute__((ext_vector_type(16))) __bf16 v16bf;
typedef __attribute__((ext_vector_type(8)))  float  v8f;

union Frag32B { uint4 u[2]; v16bf f; };

__device__ inline bf16_t to_bf16(float f) { return (bf16_t)f; }

__device__ inline v8f wmma_bf16(v16bf a, v16bf b, v8f c) {
  return __builtin_amdgcn_wmma_f32_16x16x32_bf16(
      /*neg_a=*/false, a, /*neg_b=*/false, b,
      /*c_mod=*/(short)0, c, /*reuse_a=*/false, /*reuse_b=*/false);
}

// A-matrix 16x32 bf16 fragment (row-major A, lda elements).
// Lane l<16: m=row0+l, K chunks {k0..k0+7} and {k0+16..k0+23}
// Lane l>=16: m=row0+l-16, K chunks {k0+8..15} and {k0+24..31}
__device__ inline v16bf load_frag_a(const bf16_t* A, int lda, int row0, int k0, int lane) {
  const int m  = row0 + (lane & 15);
  const int ko = k0 + ((lane & 16) ? 8 : 0);
  const bf16_t* p = A + (size_t)m * (size_t)lda + ko;
  Frag32B fr;
  fr.u[0] = *(const uint4*)(p);
  fr.u[1] = *(const uint4*)(p + 16);
  return fr.f;
}

// B-matrix 32x16 bf16 fragment where B[k][n] = W[col0+n][k0+k], W row-major [Nout, ldw].
// Lane l<16: n=col0+l, K chunks {k0..k0+7} and {k0+16..k0+23}; lane l>=16: K chunks {k0+8..15} and {k0+24..31}
__device__ inline v16bf load_frag_b(const bf16_t* W, int ldw, int col0, int k0, int lane) {
  const int n  = col0 + (lane & 15);
  const int ko = k0 + ((lane & 16) ? 8 : 0);
  const bf16_t* p = W + (size_t)n * (size_t)ldw + ko;
  Frag32B fr;
  fr.u[0] = *(const uint4*)(p);
  fr.u[1] = *(const uint4*)(p + 16);
  return fr.f;
}

// ---------------- gfx1250 async global->LDS copy (ASYNCcnt) ----------------
// GLOBAL_LOAD_ASYNC_TO_LDS_B128: vdst = LDS byte offset, vaddr = 64-bit global addr.
__device__ inline void cp_async16(const bf16_t* gptr, bf16_t* lptr) {
  unsigned lds_off = (unsigned)(unsigned long long)(uintptr_t)lptr; // generic->LDS: low 32 bits
  unsigned long long gaddr = (unsigned long long)(uintptr_t)gptr;
  asm volatile("global_load_async_to_lds_b128 %0, %1, off"
               :: "v"(lds_off), "v"(gaddr) : "memory");
}
#define ASYNC_WAIT(n) asm volatile("s_wait_asynccnt %0" :: "i"(n) : "memory")

// ---------------- cast kernel ----------------
__global__ __launch_bounds__(256)
void cast_f32_bf16_k(const float* __restrict__ src, bf16_t* __restrict__ dst, size_t n) {
  size_t i = (size_t)blockIdx.x * blockDim.x + threadIdx.x;
  size_t stride = (size_t)gridDim.x * blockDim.x;
  for (; i < n; i += stride) dst[i] = to_bf16(src[i]);
}

// ---------------- LayerNorm -> bf16 ----------------
__global__ __launch_bounds__(256)
void ln_k(const float* __restrict__ x,
          const float* __restrict__ w0, const float* __restrict__ b0,
          const float* __restrict__ w1, const float* __restrict__ b1,
          int Tsplit, bf16_t* __restrict__ out) {
  const int m = blockIdx.x;
  const int n = m % Nn;
  const float* w = (n < Tsplit) ? w0 : w1;
  const float* b = (n < Tsplit) ? b0 : b1;
  const float* xr = x + (size_t)m * Dm;
  __shared__ float red[256];
  float s = 0.f;
  for (int j = threadIdx.x; j < Dm; j += 256) s += xr[j];
  red[threadIdx.x] = s; __syncthreads();
  for (int o = 128; o > 0; o >>= 1) {
    if (threadIdx.x < o) red[threadIdx.x] += red[threadIdx.x + o];
    __syncthreads();
  }
  const float mean = red[0] / (float)Dm;
  __syncthreads();
  float vs = 0.f;
  for (int j = threadIdx.x; j < Dm; j += 256) { float d = xr[j] - mean; vs += d * d; }
  red[threadIdx.x] = vs; __syncthreads();
  for (int o = 128; o > 0; o >>= 1) {
    if (threadIdx.x < o) red[threadIdx.x] += red[threadIdx.x + o];
    __syncthreads();
  }
  const float inv = rsqrtf(red[0] / (float)Dm + 1e-5f);
  for (int j = threadIdx.x; j < Dm; j += 256)
    out[(size_t)m * Dm + j] = to_bf16((xr[j] - mean) * inv * w[j] + b[j]);
}

// ---------------- LDS-staged double-buffered bf16 WMMA GEMM: C = A @ W^T ----------------
constexpr int EPI_QKV   = 0;  // split q/k into [B,N,D] bf16, v transposed [B,H,DH,N]
constexpr int EPI_RESID = 1;  // x[m,j] += gamma[j]*(acc + bias[j])  (f32 in/out)
constexpr int EPI_GELU  = 2;  // out_bf[m,j] = gelu(acc + bias[j])   (bf16)

// Block tile 128x128, K-step 32. 8 waves in 2x4 grid; wave tile 64x32 = 8 WMMAs/step.
template<int EPI>
__global__ __launch_bounds__(256)
void gemm_bf16_k(const bf16_t* __restrict__ A,
                 const bf16_t* __restrict__ Wt, const bf16_t* __restrict__ Wf,
                 const float* __restrict__ bias_t, const float* __restrict__ bias_f,
                 int Nout, int K,
                 float* __restrict__ xio, const float* __restrict__ gamma,
                 bf16_t* __restrict__ out_bf,
                 bf16_t* __restrict__ qbv, bf16_t* __restrict__ kbv, bf16_t* __restrict__ vtv) {
  __shared__ bf16_t sA[2][128 * 32];
  __shared__ bf16_t sB[2][128 * 32];

  const int tid  = threadIdx.x;
  const int wave = tid >> 5;
  const int lane = tid & 31;
  const int row_blk = blockIdx.x * 128;
  const int col_blk = blockIdx.y * 128;
  const int wrow = (wave >> 2) * 64;   // 0 or 64
  const int wcol = (wave & 3) * 32;    // 0,32,64,96

  const bool tpart = ((row_blk % Nn) < Tt);
  const bf16_t* W    = tpart ? Wt : Wf;
  const float*  bias = tpart ? bias_t : bias_f;

  // stage one 128x32 tile of A and W into LDS buffer `buf` (4 async b128 per thread)
  auto stage = [&](int k0, int buf) {
#pragma unroll
    for (int c = 0; c < 2; ++c) {
      const int ch = tid + c * 256;           // 0..511
      const int r = ch >> 2, q = (ch & 3) * 8;
      cp_async16(A + (size_t)(row_blk + r) * K + k0 + q, &sA[buf][r * 32 + q]);
    }
#pragma unroll
    for (int c = 0; c < 2; ++c) {
      const int ch = tid + c * 256;
      const int r = ch >> 2, q = (ch & 3) * 8;
      cp_async16(W + (size_t)(col_blk + r) * K + k0 + q, &sB[buf][r * 32 + q]);
    }
    // far prefetch of the weight stream into L2 (global_prefetch_b8)
    __builtin_prefetch(W + (size_t)(col_blk + (tid & 127)) * K + k0 + 64, 0, 1);
  };

  v8f acc[4][2];
#pragma unroll
  for (int s = 0; s < 4; ++s) { acc[s][0] = v8f{}; acc[s][1] = v8f{}; }

  const int nk = K / 32;
  stage(0, 0);
  for (int i = 0; i < nk; ++i) {
    const int buf = i & 1;
    if (i + 1 < nk) { stage((i + 1) * 32, buf ^ 1); ASYNC_WAIT(4); }
    else            { ASYNC_WAIT(0); }
    __syncthreads();

    v16bf b0 = load_frag_b(sB[buf], 32, wcol,      0, lane);
    v16bf b1 = load_frag_b(sB[buf], 32, wcol + 16, 0, lane);
#pragma unroll
    for (int s = 0; s < 4; ++s) {
      v16bf a = load_frag_a(sA[buf], 32, wrow + s * 16, 0, lane);
      acc[s][0] = wmma_bf16(a, b0, acc[s][0]);
      acc[s][1] = wmma_bf16(a, b1, acc[s][1]);
    }
    __syncthreads();
  }

  // ---- epilogue ----
  const int rb = (lane & 16) ? 8 : 0;
#pragma unroll
  for (int s = 0; s < 4; ++s) {
#pragma unroll
    for (int cix = 0; cix < 2; ++cix) {
      const int col = col_blk + wcol + cix * 16 + (lane & 15);
#pragma unroll
      for (int r = 0; r < 8; ++r) {
        const int row = row_blk + wrow + s * 16 + rb + r;
        const float v = acc[s][cix][r];
        if constexpr (EPI == EPI_QKV) {
          if (col < Dm) {
            qbv[(size_t)row * Dm + col] = to_bf16(v);
          } else if (col < 2 * Dm) {
            kbv[(size_t)row * Dm + (col - Dm)] = to_bf16(v);
          } else {
            const int c  = col - 2 * Dm;
            const int hh = c / DHd, dd = c % DHd;
            const int b2 = row / Nn, n2 = row % Nn;
            vtv[(((size_t)b2 * Hh + hh) * DHd + dd) * Nn + n2] = to_bf16(v);
          }
        } else if constexpr (EPI == EPI_RESID) {
          xio[(size_t)row * Dm + col] += gamma[col] * (v + bias[col]);
        } else { // EPI_GELU (exact erf)
          const float z = v + bias[col];
          out_bf[(size_t)row * (size_t)Nout + col] =
              to_bf16(0.5f * z * (1.0f + erff(z * 0.70710678118654752f)));
        }
      }
    }
  }
}

// ---------------- attention: one block per (b, h, 16 q-rows) ----------------
// smem: 16*N f32 scores | 16*N bf16 probs | 128 f32 reduction
constexpr size_t ATTN_SMEM = (size_t)16 * Nn * 4 + (size_t)16 * Nn * 2 + 512;

__global__ __launch_bounds__(128)
void attn_k(const bf16_t* __restrict__ qb, const bf16_t* __restrict__ kb,
            const bf16_t* __restrict__ vt,
            const int* __restrict__ rel_idx, const float* __restrict__ rel_table,
            const unsigned char* __restrict__ mask,
            bf16_t* __restrict__ ob) {
  extern __shared__ __align__(16) char smem[];
  float*  sS   = (float*)smem;
  bf16_t* sP   = (bf16_t*)(smem + (size_t)16 * Nn * 4);
  float*  sred = (float*)(smem + (size_t)16 * Nn * 4 + (size_t)16 * Nn * 2);

  const int q0 = blockIdx.x * 16;
  const int hh = blockIdx.y;
  const int bb = blockIdx.z;
  const int tid  = threadIdx.x;
  const int wave = tid >> 5;    // 0..3
  const int lane = tid & 31;

  const bf16_t* qbase = qb + (size_t)bb * Nn * Dm + hh * DHd;
  const bf16_t* kbase = kb + (size_t)bb * Nn * Dm + hh * DHd;

  // ---- scores S = scale * Q K^T ----
  for (int kt = wave; kt < Nn / 16; kt += 4) {
    v8f acc = {};
    for (int d0 = 0; d0 < DHd; d0 += 32) {
      v16bf a = load_frag_a(qbase, Dm, q0,      d0, lane);
      v16bf b = load_frag_b(kbase, Dm, kt * 16, d0, lane);
      acc = wmma_bf16(a, b, acc);
    }
    const int col = kt * 16 + (lane & 15);
    const int rbv = (lane & 16) ? 8 : 0;
#pragma unroll
    for (int r = 0; r < 8; ++r) sS[(size_t)(r + rbv) * Nn + col] = acc[r] * SCALE;
  }
  __syncthreads();

  // ---- relative-position bias + key mask ----
  for (int idx = tid; idx < 16 * Nn; idx += 128) {
    const int qq = idx >> 10;
    const int kk = idx & (Nn - 1);
    float v = sS[(size_t)qq * Nn + kk] +
              rel_table[(size_t)rel_idx[(size_t)(q0 + qq) * Nn + kk] * Hh + hh];
    if (!mask[(size_t)bb * Nn + kk]) v = -__builtin_inff();
    sS[(size_t)qq * Nn + kk] = v;
  }
  __syncthreads();

  // ---- softmax over k: 8 threads per row ----
  const int row = tid >> 3;
  const int sub = tid & 7;
  float mx = -__builtin_inff();
  for (int kk = sub; kk < Nn; kk += 8) mx = fmaxf(mx, sS[(size_t)row * Nn + kk]);
  sred[tid] = mx; __syncthreads();
  if (sub == 0) {
    float m2 = sred[row * 8];
    for (int i = 1; i < 8; ++i) m2 = fmaxf(m2, sred[row * 8 + i]);
    sred[row * 8] = m2;
  }
  __syncthreads();
  const float rmax = sred[row * 8];
  __syncthreads();
  float sum = 0.f;
  for (int kk = sub; kk < Nn; kk += 8) {
    const float e = __expf(sS[(size_t)row * Nn + kk] - rmax);
    sS[(size_t)row * Nn + kk] = e;
    sum += e;
  }
  sred[tid] = sum; __syncthreads();
  if (sub == 0) {
    float s2 = 0.f;
    for (int i = 0; i < 8; ++i) s2 += sred[row * 8 + i];
    sred[row * 8] = s2;
  }
  __syncthreads();
  const float inv = 1.0f / sred[row * 8];
  for (int kk = sub; kk < Nn; kk += 8)
    sP[(size_t)row * Nn + kk] = to_bf16(sS[(size_t)row * Nn + kk] * inv);
  __syncthreads();

  // ---- O = P @ V (A-fragments from LDS via ds_load, V transposed [B,H,DH,N]) ----
  const bf16_t* vtb = vt + ((size_t)bb * Hh + hh) * DHd * Nn;
  const int d0 = wave * 16;
  v8f acc = {};
  for (int k0 = 0; k0 < Nn; k0 += 32) {
    v16bf a = load_frag_a(sP, Nn, 0, k0, lane);
    v16bf b = load_frag_b(vtb, Nn, d0, k0, lane);
    acc = wmma_bf16(a, b, acc);
  }
  const int col = hh * DHd + d0 + (lane & 15);
  const int rbv = (lane & 16) ? 8 : 0;
#pragma unroll
  for (int r = 0; r < 8; ++r) {
    const int qq = q0 + rbv + r;
    ob[((size_t)bb * Nn + qq) * Dm + col] = to_bf16(acc[r]);
  }
}

// ---------------- host driver ----------------
extern "C" void kernel_launch(void* const* d_in, const int* in_sizes, int n_in,
                              void* d_out, int out_size, void* d_ws, size_t ws_size,
                              hipStream_t stream) {
  (void)in_sizes; (void)n_in; (void)out_size; (void)ws_size;
  const float* x_in     = (const float*)d_in[0];
  const unsigned char* mask = (const unsigned char*)d_in[1];
  const int*   rel_idx  = (const int*)d_in[2];
  const float* qkv_w    = (const float*)d_in[3];
  const float* proj_w   = (const float*)d_in[4];
  const float* proj_b   = (const float*)d_in[5];
  const float* norm1_w  = (const float*)d_in[6];
  const float* norm1_b  = (const float*)d_in[7];
  const float* gamma1   = (const float*)d_in[8];
  const float* gamma2   = (const float*)d_in[9];
  const float* rel_tab  = (const float*)d_in[10];
  const float* normt_w  = (const float*)d_in[11];
  const float* normt_b  = (const float*)d_in[12];
  const float* fc1t_w   = (const float*)d_in[13];
  const float* fc1t_b   = (const float*)d_in[14];
  const float* fc2t_w   = (const float*)d_in[15];
  const float* fc2t_b   = (const float*)d_in[16];
  const float* normf_w  = (const float*)d_in[17];
  const float* normf_b  = (const float*)d_in[18];
  const float* fc1f_w   = (const float*)d_in[19];
  const float* fc1f_b   = (const float*)d_in[20];
  const float* fc2f_w   = (const float*)d_in[21];
  const float* fc2f_b   = (const float*)d_in[22];

  char* ws = (char*)d_ws;
  size_t off = 0;
  auto take = [&](size_t bytes) -> void* {
    void* p = ws + off;
    off = (off + bytes + 255) & ~(size_t)255;
    return p;
  };
  float*  xbuf  = (float*) take((size_t)Mrows * Dm * 4);
  bf16_t* hbuf  = (bf16_t*)take((size_t)Mrows * Dm * 2);
  bf16_t* qbuf  = (bf16_t*)take((size_t)Mrows * Dm * 2);
  bf16_t* kbuf  = (bf16_t*)take((size_t)Mrows * Dm * 2);
  bf16_t* vtbuf = (bf16_t*)take((size_t)Mrows * Dm * 2);
  bf16_t* obuf  = (bf16_t*)take((size_t)Mrows * Dm * 2);
  bf16_t* z1buf = (bf16_t*)take((size_t)Mrows * HIDd * 2);
  bf16_t* wqkv  = (bf16_t*)take((size_t)3 * Dm * Dm * 2);
  bf16_t* wproj = (bf16_t*)take((size_t)Dm * Dm * 2);
  bf16_t* wfc1t = (bf16_t*)take((size_t)HIDd * Dm * 2);
  bf16_t* wfc1f = (bf16_t*)take((size_t)HIDd * Dm * 2);
  bf16_t* wfc2t = (bf16_t*)take((size_t)Dm * HIDd * 2);
  bf16_t* wfc2f = (bf16_t*)take((size_t)Dm * HIDd * 2);

  (void)hipFuncSetAttribute((const void*)attn_k,
                            hipFuncAttributeMaxDynamicSharedMemorySize,
                            (int)ATTN_SMEM);

  hipMemcpyAsync(xbuf, x_in, (size_t)Mrows * Dm * 4, hipMemcpyDeviceToDevice, stream);

  auto cast_launch = [&](const float* s, bf16_t* d, size_t n) {
    cast_f32_bf16_k<<<2048, 256, 0, stream>>>(s, d, n);
  };

  const dim3 blk(256);
  for (int l = 0; l < Lq; ++l) {
    cast_launch(qkv_w  + (size_t)l * 3 * Dm * Dm, wqkv,  (size_t)3 * Dm * Dm);
    cast_launch(proj_w + (size_t)l * Dm * Dm,     wproj, (size_t)Dm * Dm);
    cast_launch(fc1t_w + (size_t)l * HIDd * Dm,   wfc1t, (size_t)HIDd * Dm);
    cast_launch(fc1f_w + (size_t)l * HIDd * Dm,   wfc1f, (size_t)HIDd * Dm);
    cast_launch(fc2t_w + (size_t)l * Dm * HIDd,   wfc2t, (size_t)Dm * HIDd);
    cast_launch(fc2f_w + (size_t)l * Dm * HIDd,   wfc2f, (size_t)Dm * HIDd);

    // h = LN(x)
    ln_k<<<Mrows, blk, 0, stream>>>(xbuf,
        norm1_w + (size_t)l * Dm, norm1_b + (size_t)l * Dm,
        norm1_w + (size_t)l * Dm, norm1_b + (size_t)l * Dm, Nn, hbuf);

    // qkv = h @ qkv_w^T  (split into q/k row-major, v transposed)
    gemm_bf16_k<EPI_QKV><<<dim3(Mrows / 128, (3 * Dm) / 128), blk, 0, stream>>>(
        hbuf, wqkv, wqkv, nullptr, nullptr, 3 * Dm, Dm,
        nullptr, nullptr, nullptr, qbuf, kbuf, vtbuf);

    // attention
    attn_k<<<dim3(Nn / 16, Hh, Bb), dim3(128), ATTN_SMEM, stream>>>(
        qbuf, kbuf, vtbuf, rel_idx, rel_tab + (size_t)l * Rr * Hh, mask, obuf);

    // x += gamma1 * (o @ proj_w^T + proj_b)
    gemm_bf16_k<EPI_RESID><<<dim3(Mrows / 128, Dm / 128), blk, 0, stream>>>(
        obuf, wproj, wproj, proj_b + (size_t)l * Dm, proj_b + (size_t)l * Dm, Dm, Dm,
        xbuf, gamma1 + (size_t)l * Dm, nullptr, nullptr, nullptr, nullptr);

    // h = LN_split(x) (t tokens vs f tokens)
    ln_k<<<Mrows, blk, 0, stream>>>(xbuf,
        normt_w + (size_t)l * Dm, normt_b + (size_t)l * Dm,
        normf_w + (size_t)l * Dm, normf_b + (size_t)l * Dm, Tt, hbuf);

    // z1 = gelu(h @ fc1^T + b1)   (branch weights per 128-row block)
    gemm_bf16_k<EPI_GELU><<<dim3(Mrows / 128, HIDd / 128), blk, 0, stream>>>(
        hbuf, wfc1t, wfc1f, fc1t_b + (size_t)l * HIDd, fc1f_b + (size_t)l * HIDd,
        HIDd, Dm, nullptr, nullptr, z1buf, nullptr, nullptr, nullptr);

    // x += gamma2 * (z1 @ fc2^T + b2)
    gemm_bf16_k<EPI_RESID><<<dim3(Mrows / 128, Dm / 128), blk, 0, stream>>>(
        z1buf, wfc2t, wfc2f, fc2t_b + (size_t)l * Dm, fc2f_b + (size_t)l * Dm,
        Dm, HIDd, xbuf, gamma2 + (size_t)l * Dm, nullptr, nullptr, nullptr, nullptr);
  }

  hipMemcpyAsync(d_out, xbuf, (size_t)Mrows * Dm * 4, hipMemcpyDeviceToDevice, stream);
}
